// SetTransformer_6236292514218
// MI455X (gfx1250) — compile-verified, matrix operand order checked by reading, not run on hardware
//
#include <hip/hip_runtime.h>
#include <hip/hip_bf16.h>

#define DV    128
#define NHEAD 4
#define HDIM  32
#define BSZ   512
#define NSEQ  256
#define NIND  32
#define NSEED 1

typedef __attribute__((ext_vector_type(16))) __bf16       v16bf;
typedef __attribute__((ext_vector_type(8)))  float        v8f;
typedef __attribute__((ext_vector_type(4)))  unsigned int v4u;

union Frag16 {
  v16bf v;
  v4u   q[2];
  unsigned int u[8];
};

__device__ __forceinline__ unsigned short f2bf(float f) {
  unsigned int u = __float_as_uint(f);
  u += 0x7FFFu + ((u >> 16) & 1u);          // round-to-nearest-even
  return (unsigned short)(u >> 16);
}
__device__ __forceinline__ float bf2f(unsigned short h) {
  return __uint_as_float(((unsigned int)h) << 16);
}

// -----------------------------------------------------------------------------
// Y[M x 128] = epilogue( X[M x 128] @ W[128 x 128] + bias )
//   mode 0: Y = X@W + b                   (bf16 out)       [q/k/v projections]
//   mode 1: Y = X + relu(X@W + b)         (bf16 out)       [MAB output block]
//   mode 2: Y = X@W + b                   (f32 out)        [branch fc]
// src_mod > 0: A row r reads X[r % src_mod]  (broadcast of I / S over batch)
// 4 waves/block; each wave computes a 16x128 strip with 8 f32 WMMA accumulators.
// W is transposed to bf16 in LDS so B fragments are contiguous ds_load_b128s.
// -----------------------------------------------------------------------------
__global__ void __launch_bounds__(128)
gemm128_kernel(const unsigned short* __restrict__ X,
               const float* __restrict__ W,
               const float* __restrict__ bias,
               unsigned short* __restrict__ Ybf,
               float* __restrict__ Yf,
               int M, int src_mod, int mode)
{
  __shared__ unsigned short Wt[DV * DV];   // Wt[n][k] = bf16(W[k][n])  (32 KB)
  __shared__ float bsh[DV];
  const int tid = threadIdx.x;
  for (int i = tid; i < DV * DV; i += 128) {
    int k = i >> 7, n = i & (DV - 1);
    Wt[n * DV + k] = f2bf(W[i]);
  }
  bsh[tid] = bias[tid];
  __syncthreads();

  const int wave = tid >> 5;
  const int lane = tid & 31;
  const int g = lane >> 4;      // K-half select per WMMA 16-bit layout
  const int m = lane & 15;
  const long rowBase = (long)blockIdx.x * 64 + wave * 16;
  if (rowBase >= M) return;
  const long arow = rowBase + m;
  const long srow = (src_mod > 0) ? (arow % src_mod) : arow;

  v8f zero = {};
  v8f acc[8];
#pragma unroll
  for (int t = 0; t < 8; ++t) acc[t] = zero;

#pragma unroll
  for (int kk = 0; kk < 4; ++kk) {
    const int kb = kk * 32;
    Frag16 a;
    const v4u* pa = (const v4u*)(X + srow * DV + kb + g * 8);
    a.q[0] = pa[0];             // K = kb + g*8 .. +7
    a.q[1] = pa[2];             // K = kb + 16 + g*8 .. +7
#pragma unroll
    for (int t = 0; t < 8; ++t) {
      Frag16 bfr;
      const v4u* pb = (const v4u*)(&Wt[(t * 16 + m) * DV + kb + g * 8]);
      bfr.q[0] = pb[0];
      bfr.q[1] = pb[2];
      acc[t] = __builtin_amdgcn_wmma_f32_16x16x32_bf16(
          false, a.v, false, bfr.v, (short)0, acc[t], false, false);
    }
  }

#pragma unroll
  for (int t = 0; t < 8; ++t) {
    const int col = t * 16 + m;
    const float bv = bsh[col];
#pragma unroll
    for (int r = 0; r < 8; ++r) {
      const long orow = rowBase + r + g * 8;     // D-layout: lanes 16-31 -> M+8
      float v = acc[t][r] + bv;
      if (mode == 1) {
        float xr = bf2f(X[orow * DV + col]);
        v = xr + (v > 0.0f ? v : 0.0f);
      }
      if (mode == 2) Yf[orow * DV + col] = v;
      else           Ybf[orow * DV + col] = f2bf(v);
    }
  }
}

// -----------------------------------------------------------------------------
// Attention for one (batch, head, 16-row q tile); one wave per block.
// scores = softmax( (qh @ khT) / sqrt(DV) );  O = q + scores @ vh   (in-place OK)
// d=32 == WMMA K exactly: one v_wmma per 16x16 score tile.
// -----------------------------------------------------------------------------
__global__ void __launch_bounds__(32)
attn_kernel(const unsigned short* __restrict__ Qb,   // [B*nq][128] bf16
            const unsigned short* __restrict__ Kb,   // [B*nk][128]
            const unsigned short* __restrict__ Vb,   // [B*nk][128]
            unsigned short* __restrict__ Ob,         // [B*nq][128] (may == Qb)
            int nq, int nk)
{
  extern __shared__ char smem[];
  const int kpad = (nk + 31) & ~31;
  float* sc          = (float*)smem;                         // [16][kpad] f32
  unsigned short* pr = (unsigned short*)(sc + 16 * kpad);    // [16][kpad] bf16
  unsigned short* vt = pr + 16 * kpad;                       // [32][kpad] V^T bf16

  const int qtiles = (nq + 15) >> 4;
  int bid = blockIdx.x;
  const int qt = bid % qtiles; bid /= qtiles;
  const int h  = bid % NHEAD;
  const int b  = bid / NHEAD;

  const int lane = threadIdx.x;
  const int g = lane >> 4;
  const int m = lane & 15;

  // A fragment: 16x32 q tile (rows beyond nq zero-padded)
  Frag16 qa;
  const int qrow = qt * 16 + m;
  if (qrow < nq) {
    const v4u* p = (const v4u*)(Qb + ((long)b * nq + qrow) * DV + h * HDIM + g * 8);
    qa.q[0] = p[0];
    qa.q[1] = p[2];
  } else {
#pragma unroll
    for (int i = 0; i < 8; ++i) qa.u[i] = 0u;
  }

  // stage V^T into LDS, zero-padded to kpad
  for (int j = lane; j < kpad; j += 32) {
    if (j < nk) {
      const unsigned short* src = Vb + ((long)b * nk + j) * DV + h * HDIM;
#pragma unroll
      for (int c = 0; c < HDIM; ++c) vt[c * kpad + j] = src[c];
    } else {
#pragma unroll
      for (int c = 0; c < HDIM; ++c) vt[c * kpad + j] = 0;
    }
  }

  // scores = qh @ kh^T  (B column k == key row k: contiguous loads)
  const float scale = 0.088388347648318447f;   // 1/sqrt(DV=128)
  const int ktiles = (nk + 15) >> 4;
  for (int kt = 0; kt < ktiles; ++kt) {
    Frag16 kf;
    const int krow = kt * 16 + m;
    if (krow < nk) {
      const v4u* p = (const v4u*)(Kb + ((long)b * nk + krow) * DV + h * HDIM + g * 8);
      kf.q[0] = p[0];
      kf.q[1] = p[2];
    } else {
#pragma unroll
      for (int i = 0; i < 8; ++i) kf.u[i] = 0u;
    }
    v8f s = {};
    s = __builtin_amdgcn_wmma_f32_16x16x32_bf16(false, qa.v, false, kf.v,
                                                (short)0, s, false, false);
#pragma unroll
    for (int r = 0; r < 8; ++r)
      sc[(r + g * 8) * kpad + kt * 16 + m] = s[r] * scale;
  }
  __syncthreads();

  // softmax: one row per lane (lanes 0-15), fp32, then quantize probs to bf16
  if (lane < 16) {
    float mx = -1e30f;
    for (int j = 0; j < nk; ++j) mx = fmaxf(mx, sc[lane * kpad + j]);
    float sum = 0.0f;
    for (int j = 0; j < nk; ++j) {
      float e = __expf(sc[lane * kpad + j] - mx);
      sc[lane * kpad + j] = e;
      sum += e;
    }
    float inv = 1.0f / sum;
    for (int j = 0; j < kpad; ++j)
      pr[lane * kpad + j] = (j < nk) ? f2bf(sc[lane * kpad + j] * inv)
                                     : (unsigned short)0;
  }
  __syncthreads();

  // O = probs @ V  (K = kpad, two 16-col N tiles)
  v8f o0 = {}, o1 = {};
  for (int kc = 0; kc < kpad; kc += 32) {
    Frag16 pa, b0, b1;
    const v4u* pp = (const v4u*)(pr + m * kpad + kc + g * 8);
    pa.q[0] = pp[0]; pa.q[1] = pp[2];
    const v4u* p0 = (const v4u*)(vt + m * kpad + kc + g * 8);
    b0.q[0] = p0[0]; b0.q[1] = p0[2];
    const v4u* p1 = (const v4u*)(vt + (16 + m) * kpad + kc + g * 8);
    b1.q[0] = p1[0]; b1.q[1] = p1[2];
    o0 = __builtin_amdgcn_wmma_f32_16x16x32_bf16(false, pa.v, false, b0.v,
                                                 (short)0, o0, false, false);
    o1 = __builtin_amdgcn_wmma_f32_16x16x32_bf16(false, pa.v, false, b1.v,
                                                 (short)0, o1, false, false);
  }

  // O = q + attn (residual); all reads precede these stores in program order
#pragma unroll
  for (int r = 0; r < 8; ++r) {
    const int qr = qt * 16 + r + g * 8;
    if (qr < nq) {
      const long base = ((long)b * nq + qr) * DV + h * HDIM;
      Ob[base + m]      = f2bf(bf2f(Qb[base + m])      + o0[r]);
      Ob[base + 16 + m] = f2bf(bf2f(Qb[base + 16 + m]) + o1[r]);
    }
  }
}

// -----------------------------------------------------------------------------
__global__ void __launch_bounds__(256)
gather_embed_kernel(const int* __restrict__ idx, const float* __restrict__ emb,
                    unsigned short* __restrict__ X, long nelem)
{
  long e = (long)blockIdx.x * 256 + threadIdx.x;
  if (e >= nelem) return;
  long t = e >> 7;
  int  c = (int)(e & (DV - 1));
  X[e] = f2bf(emb[idx[t] * DV + c]);
}

__global__ void __launch_bounds__(256)
cvt_bf16_kernel(const float* __restrict__ src, unsigned short* __restrict__ dst, int n)
{
  int i = blockIdx.x * 256 + threadIdx.x;
  if (i < n) dst[i] = f2bf(src[i]);
}

// x = p*e ; h = leaky_relu(x@W1+b1) ; out = sigmoid(h@W2+b2)
__global__ void __launch_bounds__(64)
head_kernel(const float* __restrict__ P, const float* __restrict__ E,
            const float* __restrict__ W1, const float* __restrict__ b1,
            const float* __restrict__ W2, const float* __restrict__ b2,
            float* __restrict__ out)
{
  __shared__ float xs[DV];
  __shared__ float hs[64];
  const int b = blockIdx.x;
  const int t = threadIdx.x;
  for (int i = t; i < DV; i += 64) xs[i] = P[b * DV + i] * E[b * DV + i];
  __syncthreads();
  float acc = b1[t];
  for (int k = 0; k < DV; ++k) acc += xs[k] * W1[k * 64 + t];
  float h = acc > 0.0f ? acc : 0.01f * acc;   // jax leaky_relu default slope
  hs[t] = h * W2[t];
  __syncthreads();
  for (int s = 32; s > 0; s >>= 1) {
    if (t < s) hs[t] += hs[t + s];
    __syncthreads();
  }
  if (t == 0) out[b] = 1.0f / (1.0f + __expf(-(hs[0] + b2[0])));
}

// -----------------------------------------------------------------------------
struct Lin { const float* W; const float* b; };
struct Mab { Lin q, k, v, o; };
struct IsabP { const float* I; Mab m0, m1; };
struct BranchP { IsabP i0, i1; const float* S; Mab pma, s0, s1; Lin fc; };

extern "C" void kernel_launch(void* const* d_in, const int* in_sizes, int n_in,
                              void* d_out, int out_size, void* d_ws, size_t ws_size,
                              hipStream_t stream)
{
  (void)in_sizes; (void)n_in; (void)out_size; (void)ws_size;
  int ix = 0;
  const int*   para_idx = (const int*)d_in[ix++];
  const int*   epi_idx  = (const int*)d_in[ix++];
  const float* emb      = (const float*)d_in[ix++];
  auto lin   = [&]() { Lin L; L.W = (const float*)d_in[ix++]; L.b = (const float*)d_in[ix++]; return L; };
  auto mabp  = [&]() { Mab m; m.q = lin(); m.k = lin(); m.v = lin(); m.o = lin(); return m; };
  auto isabp = [&]() { IsabP s; s.I = (const float*)d_in[ix++]; s.m0 = mabp(); s.m1 = mabp(); return s; };
  auto brp   = [&]() {
    BranchP bb; bb.i0 = isabp(); bb.i1 = isabp();
    bb.S = (const float*)d_in[ix++];
    bb.pma = mabp(); bb.s0 = mabp(); bb.s1 = mabp(); bb.fc = lin(); return bb;
  };
  BranchP Pb = brp();
  BranchP Eb = brp();
  Lin fc1 = lin(), fc2 = lin();

  char* ws = (char*)d_ws;
  const size_t BIG = (size_t)BSZ * NSEQ * DV * sizeof(unsigned short);  // 32 MB
  unsigned short* bufX = (unsigned short*)(ws);
  unsigned short* bufQ = (unsigned short*)(ws + 1 * BIG);  // also attention O (in-place)
  unsigned short* bufK = (unsigned short*)(ws + 2 * BIG);
  unsigned short* bufV = (unsigned short*)(ws + 3 * BIG);
  unsigned short* bufH = (unsigned short*)(ws + 4 * BIG);  // [BSZ*NIND*DV] max
  unsigned short* bufI = (unsigned short*)(ws + 4 * BIG + (size_t)BSZ * NIND * DV * 2);
  float* outP = (float*)(ws + 4 * BIG + (size_t)BSZ * NIND * DV * 2 + 65536);
  float* outE = outP + BSZ * DV;

  auto gemm = [&](const unsigned short* X, Lin L, unsigned short* Ybf, float* Yf,
                  int M, int src_mod, int mode) {
    gemm128_kernel<<<dim3(M / 64), dim3(128), 0, stream>>>(X, L.W, L.b, Ybf, Yf,
                                                           M, src_mod, mode);
  };
  auto attn = [&](const unsigned short* Q, const unsigned short* K,
                  const unsigned short* V, unsigned short* O, int nq, int nk) {
    int kpad = (nk + 31) & ~31;
    size_t sh = (size_t)kpad * (16 * 4 + 16 * 2 + 32 * 2);   // sc + pr + vt
    int qtiles = (nq + 15) / 16;
    attn_kernel<<<dim3(BSZ * NHEAD * qtiles), dim3(32), sh, stream>>>(Q, K, V, O, nq, nk);
  };
  auto mab = [&](const Mab& p, const unsigned short* Qin, int nq, int srcModQ,
                 const unsigned short* Kin, int nk, unsigned short* out) {
    gemm(Qin, p.q, bufQ, nullptr, BSZ * nq, srcModQ, 0);
    gemm(Kin, p.k, bufK, nullptr, BSZ * nk, 0, 0);
    gemm(Kin, p.v, bufV, nullptr, BSZ * nk, 0, 0);
    attn(bufQ, bufK, bufV, bufQ, nq, nk);            // O = q + attn, in place
    gemm(bufQ, p.o, out, nullptr, BSZ * nq, 0, 1);   // out = O + relu(O@Wo+bo)
  };
  auto cvt = [&](const float* src, unsigned short* dst, int n) {
    cvt_bf16_kernel<<<dim3((n + 255) / 256), dim3(256), 0, stream>>>(src, dst, n);
  };
  auto isab = [&](const IsabP& s) {
    cvt(s.I, bufI, NIND * DV);
    mab(s.m0, bufI, NIND, NIND, bufX, NSEQ, bufH);   // Hm = MAB(I, X)
    mab(s.m1, bufX, NSEQ, 0, bufH, NIND, bufX);      // X  = MAB(X, Hm)
  };
  auto branch = [&](const BranchP& bp, const int* tok, float* outv) {
    long ne = (long)BSZ * NSEQ * DV;
    gather_embed_kernel<<<dim3((unsigned)((ne + 255) / 256)), dim3(256), 0, stream>>>(
        tok, emb, bufX, ne);
    isab(bp.i0);
    isab(bp.i1);
    cvt(bp.S, bufI, NSEED * DV);
    mab(bp.pma, bufI, NSEED, NSEED, bufX, NSEQ, bufH);  // PMA
    mab(bp.s0, bufH, NSEED, 0, bufH, NSEED, bufH);      // SAB0
    mab(bp.s1, bufH, NSEED, 0, bufH, NSEED, bufH);      // SAB1
    gemm(bufH, bp.fc, nullptr, outv, BSZ * NSEED, 0, 2); // fc, mean over NS=1
  };

  branch(Pb, para_idx, outP);
  branch(Eb, epi_idx, outE);
  head_kernel<<<dim3(BSZ), dim3(64), 0, stream>>>(outP, outE, fc1.W, fc1.b,
                                                  fc2.W, fc2.b, (float*)d_out);
}
